// SimpleRNN_42511586296557
// MI455X (gfx1250) — matrix operand order repaired and checked by reading, not compiled, not run on hardware
//
#include <hip/hip_runtime.h>
#include <math.h>

// ---------------------------------------------------------------------------
// CDNA5 (gfx1250) wave32 WMMA implementation of CNN->2-layer LSTM, batch 1.
//   * bf16 weight copies (halves traffic, recurrent set ~226MB vs 192MB L2)
//   * time-parallel: conv/pool, X = feat@inp_w^T, Gih0 = X@w_ih0^T (WMMA GEMM)
//   * sequential: 3 WMMA mat-vecs + LSTM cells per step, software-pipelined
//     K loops (register multi-buffering) since only ~768 waves exist to
//     hide latency -> ILP must keep many b128 loads in flight per wave.
// ---------------------------------------------------------------------------

#define HD   3072
#define HD4  12288
#define TT   64

typedef __bf16 bf16_t;
typedef __attribute__((ext_vector_type(8)))  __bf16 v8bf;
typedef __attribute__((ext_vector_type(16))) __bf16 v16bf;
typedef __attribute__((ext_vector_type(8)))  float  v8f;

__device__ __forceinline__ unsigned short f2bf(float f) {
    unsigned int u = __float_as_uint(f);
    unsigned int r = u + 0x7FFFu + ((u >> 16) & 1u);   // round-to-nearest-even
    return (unsigned short)(r >> 16);
}

// A fragment: 16x32 bf16 tile of a row-major MxK matrix.
// lane L: m = L&15, kh = L>>4; holds 16 consecutive K at k0+kh*16
__device__ __forceinline__ v16bf load_a_frag(const unsigned short* A, int lda,
                                             int m0, int k0, int lane) {
    int m  = lane & 15;
    int kh = lane >> 4;
    const unsigned short* base = A + (size_t)(m0 + m) * lda + k0 + kh * 16;
    v8bf c0 = *reinterpret_cast<const v8bf*>(base);
    v8bf c1 = *reinterpret_cast<const v8bf*>(base + 8);
    return __builtin_shufflevector(c0, c1, 0,1,2,3,4,5,6,7,8,9,10,11,12,13,14,15);
}

// B fragment: 32x16 bf16 tile where B[k][n] = W[n0+n][k0+k]  (W row-major NxK).
// lane L: n = L&15, kh = L>>4; holds 16 consecutive K at k0+kh*16.
__device__ __forceinline__ v16bf load_b_frag(const unsigned short* W, int ldw,
                                             int n0, int k0, int lane) {
    int n  = lane & 15;
    int kh = lane >> 4;
    const unsigned short* base = W + (size_t)(n0 + n) * ldw + k0 + kh * 16;
    return *reinterpret_cast<const v16bf*>(base);
}

// Non-temporal variant for single-use (streamed-once) weight matrices so they
// do not evict the 64x-reused recurrent weights from the 192MB L2.
__device__ __forceinline__ v16bf load_b_frag_nt(const unsigned short* W, int ldw,
                                                int n0, int k0, int lane) {
    int n  = lane & 15;
    int kh = lane >> 4;
    const v16bf* base = reinterpret_cast<const v16bf*>(
        W + (size_t)(n0 + n) * ldw + k0 + kh * 16);
    return __builtin_nontemporal_load(base);
}

// A fragment for mat-vec: row 0 = h[k0..k0+31], rows 1..15 zero.
__device__ __forceinline__ v16bf load_vec_frag(const unsigned short* h, int k0, int lane) {
    int m  = lane & 15;
    int kh = lane >> 4;
    v16bf a;
    if (m == 0) {
        const unsigned short* base = h + k0 + kh * 16;
        v8bf c0 = *reinterpret_cast<const v8bf*>(base);
        v8bf c1 = *reinterpret_cast<const v8bf*>(base + 8);
        a = __builtin_shufflevector(c0, c1, 0,1,2,3,4,5,6,7,8,9,10,11,12,13,14,15);
    } else {
#pragma unroll
        for (int i = 0; i < 16; ++i) a[i] = bf16_t(0.0f);
    }
    return a;
}

#define WMMA_BF16(a, b, c) \
    __builtin_amdgcn_wmma_f32_16x16x32_bf16(false, (a), false, (b), (short)0, (c), false, false)

// ------------------------- fp32 -> bf16 conversion -------------------------
__global__ void cvt_bf16_kernel(const float* __restrict__ src,
                                unsigned short* __restrict__ dst, size_t n) {
    size_t i = (size_t)blockIdx.x * blockDim.x + threadIdx.x;
    size_t stride = (size_t)gridDim.x * blockDim.x;
    for (; i < n; i += stride) dst[i] = f2bf(src[i]);
}

// -------------------- conv3x3(p1) + relu + maxpool3x3 s2 p1 ----------------
// input layout (1,1,64,64,T): elem (y,x,t) at (y*64+x)*64 + t
// feat layout: (T, 3*32*32) row-major, bf16
__global__ void conv_pool_kernel(const float* __restrict__ in,
                                 const float* __restrict__ cw,
                                 const float* __restrict__ cb,
                                 unsigned short* __restrict__ feat) {
    int idx = blockIdx.x * blockDim.x + threadIdx.x;
    const int TOT = TT * 3 * 32 * 32;
    if (idx >= TOT) return;
    int px = idx & 31;
    int py = (idx >> 5) & 31;
    int c  = (idx >> 10) % 3;
    int t  = idx / (3 * 1024);
    float w[9];
#pragma unroll
    for (int i = 0; i < 9; ++i) w[i] = cw[c * 9 + i];
    float bias = cb[c];
    float m = -1e30f;
    for (int wy = 2 * py - 1; wy <= 2 * py + 1; ++wy) {
        if (wy < 0 || wy > 63) continue;
        for (int wx = 2 * px - 1; wx <= 2 * px + 1; ++wx) {
            if (wx < 0 || wx > 63) continue;
            float s = bias;
#pragma unroll
            for (int ky = 0; ky < 3; ++ky) {
                int iy = wy + ky - 1;
                if (iy < 0 || iy > 63) continue;
#pragma unroll
                for (int kx = 0; kx < 3; ++kx) {
                    int ix = wx + kx - 1;
                    if (ix < 0 || ix > 63) continue;
                    s += in[((size_t)(iy * 64 + ix)) * TT + t] * w[ky * 3 + kx];
                }
            }
            m = fmaxf(m, s);
        }
    }
    feat[idx] = f2bf(fmaxf(0.0f, m));
}

// ----------------- batched GEMM: out[m][n] = A[m]·W[n] + b1[n]+b2[n] -------
// A: MxK bf16 row-major; W: NxK bf16 row-major (streamed once -> NT loads).
// 4 waves / block, one 16x16 tile per wave; double-buffered K loop.
__global__ void gemm_bf16_kernel(const unsigned short* __restrict__ A, int lda,
                                 const unsigned short* __restrict__ W, int ldw,
                                 const float* __restrict__ b1,
                                 const float* __restrict__ b2,
                                 float* __restrict__ outF,
                                 unsigned short* __restrict__ outBf,
                                 int N, int K) {
    int lane  = threadIdx.x & 31;
    int wave  = threadIdx.x >> 5;
    int ntile = blockIdx.x * 4 + wave;
    int n0    = ntile * 16;
    int m0    = blockIdx.y * 16;
    if (n0 >= N) return;
    v8f acc = {};
    v16bf a0 = load_a_frag(A, lda, m0, 0, lane);
    v16bf b0 = load_b_frag_nt(W, ldw, n0, 0, lane);
    for (int k0 = 32; k0 < K; k0 += 32) {
        v16bf a1 = load_a_frag(A, lda, m0, k0, lane);
        v16bf b1v = load_b_frag_nt(W, ldw, n0, k0, lane);
        acc = WMMA_BF16(a0, b0, acc);
        a0 = a1; b0 = b1v;
    }
    acc = WMMA_BF16(a0, b0, acc);

    int n  = n0 + (lane & 15);
    int mh = lane >> 4;
    float bias = (b1 ? b1[n] : 0.0f) + (b2 ? b2[n] : 0.0f);
#pragma unroll
    for (int r = 0; r < 8; ++r) {
        int row = m0 + mh * 8 + r;
        float v = acc[r] + bias;
        if (outF) outF[(size_t)row * N + n] = v;
        else      outBf[(size_t)row * N + n] = f2bf(v);
    }
}

// ---------- recurrent mat-vec: g[n] = add[n] + h·W[n]  (triple-buffered) ---
// Only ~768 waves exist -> keep >=8 b128 loads in flight per wave (depth-3
// register pipeline) + DEV-scope prefetch to pull HBM lines into GL2 early.
__global__ void matvec_single_kernel(const unsigned short* __restrict__ h,
                                     const unsigned short* __restrict__ W,
                                     const float* __restrict__ addv,
                                     float* __restrict__ g, int N, int K) {
    int lane  = threadIdx.x & 31;
    int wave  = threadIdx.x >> 5;
    int ntile = blockIdx.x * 4 + wave;
    int n0    = ntile * 16;
    if (n0 >= N) return;
    const unsigned short* wrow = W + (size_t)(n0 + (lane & 15)) * K;

    v8f acc = {};
    v16bf a0 = load_vec_frag(h, 0, lane);
    v16bf b0 = load_b_frag(W, K, n0, 0, lane);
    v16bf a1 = load_vec_frag(h, 32, lane);
    v16bf b1 = load_b_frag(W, K, n0, 32, lane);
    for (int k0 = 64; k0 < K; k0 += 32) {
        __builtin_prefetch(wrow + k0 + 256, 0, 1);   // ~8 tiles ahead -> GL2
        v16bf a2 = load_vec_frag(h, k0, lane);
        v16bf b2 = load_b_frag(W, K, n0, k0, lane);
        acc = WMMA_BF16(a0, b0, acc);
        a0 = a1; b0 = b1;
        a1 = a2; b1 = b2;
    }
    acc = WMMA_BF16(a0, b0, acc);
    acc = WMMA_BF16(a1, b1, acc);

    if (lane < 16) {             // D row M=0 lives in acc[0] of lanes 0..15
        int n = n0 + lane;
        g[n] = acc[0] + (addv ? addv[n] : 0.0f);
    }
}

// ---- dual mat-vec: g[n] = b1[n]+b2[n] + hA·W1[n] + hB·W2[n] (2 streams,
//      double-buffered each -> 8 loads in flight) ----
__global__ void matvec_dual_kernel(const unsigned short* __restrict__ hA,
                                   const unsigned short* __restrict__ W1,
                                   const unsigned short* __restrict__ hB,
                                   const unsigned short* __restrict__ W2,
                                   const float* __restrict__ b1,
                                   const float* __restrict__ b2,
                                   float* __restrict__ g, int N, int K) {
    int lane  = threadIdx.x & 31;
    int wave  = threadIdx.x >> 5;
    int ntile = blockIdx.x * 4 + wave;
    int n0    = ntile * 16;
    if (n0 >= N) return;
    const unsigned short* w1row = W1 + (size_t)(n0 + (lane & 15)) * K;
    const unsigned short* w2row = W2 + (size_t)(n0 + (lane & 15)) * K;

    v8f acc = {};
    v16bf xa0 = load_vec_frag(hA, 0, lane);
    v16bf xb0 = load_b_frag(W1, K, n0, 0, lane);
    v16bf ya0 = load_vec_frag(hB, 0, lane);
    v16bf yb0 = load_b_frag(W2, K, n0, 0, lane);
    for (int k0 = 32; k0 < K; k0 += 32) {
        __builtin_prefetch(w1row + k0 + 224, 0, 1);
        __builtin_prefetch(w2row + k0 + 224, 0, 1);
        v16bf xa1 = load_vec_frag(hA, k0, lane);
        v16bf xb1 = load_b_frag(W1, K, n0, k0, lane);
        v16bf ya1 = load_vec_frag(hB, k0, lane);
        v16bf yb1 = load_b_frag(W2, K, n0, k0, lane);
        acc = WMMA_BF16(xa0, xb0, acc);
        acc = WMMA_BF16(ya0, yb0, acc);
        xa0 = xa1; xb0 = xb1; ya0 = ya1; yb0 = yb1;
    }
    acc = WMMA_BF16(xa0, xb0, acc);
    acc = WMMA_BF16(ya0, yb0, acc);

    if (lane < 16) {
        int n = n0 + lane;
        g[n] = acc[0] + (b1 ? b1[n] : 0.0f) + (b2 ? b2[n] : 0.0f);
    }
}

// --------------------------- LSTM cell (elementwise) -----------------------
__global__ void lstm_cell_kernel(const float* __restrict__ g,
                                 float* __restrict__ c,
                                 float* __restrict__ hf,
                                 unsigned short* __restrict__ hbf) {
    int n = blockIdx.x * blockDim.x + threadIdx.x;
    if (n >= HD) return;
    float i  = 1.0f / (1.0f + expf(-g[n]));
    float f  = 1.0f / (1.0f + expf(-g[HD + n]));
    float gg = tanhf(g[2 * HD + n]);
    float o  = 1.0f / (1.0f + expf(-g[3 * HD + n]));
    float cv = f * c[n] + i * gg;
    c[n] = cv;
    float hv = o * tanhf(cv);
    hf[n]  = hv;
    hbf[n] = f2bf(hv);
}

__global__ void init_state_kernel(float* c1, float* c2, float* h1f, float* h2f,
                                  unsigned short* h1b, unsigned short* h2b) {
    int n = blockIdx.x * blockDim.x + threadIdx.x;
    if (n >= HD) return;
    c1[n] = 0.f; c2[n] = 0.f; h1f[n] = 0.f; h2f[n] = 0.f;
    h1b[n] = 0;  h2b[n] = 0;
}

// ------------------------- final projection (scalar) -----------------------
__global__ void outdot_kernel(const float* __restrict__ h,
                              const float* __restrict__ w,
                              const float* __restrict__ b,
                              float* __restrict__ out, int t) {
    __shared__ float red[256];
    float s = 0.f;
    for (int i = threadIdx.x; i < HD; i += 256) s += h[i] * w[i];
    red[threadIdx.x] = s;
    __syncthreads();
    for (int off = 128; off > 0; off >>= 1) {
        if (threadIdx.x < off) red[threadIdx.x] += red[threadIdx.x + off];
        __syncthreads();
    }
    if (threadIdx.x == 0) out[t] = red[0] + b[0];
}

// ---------------------------------------------------------------------------
extern "C" void kernel_launch(void* const* d_in, const int* in_sizes, int n_in,
                              void* d_out, int out_size, void* d_ws, size_t ws_size,
                              hipStream_t stream) {
    const float* inp    = (const float*)d_in[0];
    const float* conv_w = (const float*)d_in[1];
    const float* conv_b = (const float*)d_in[2];
    const float* inp_w  = (const float*)d_in[3];
    const float* inp_b  = (const float*)d_in[4];
    const float* w_ih0  = (const float*)d_in[5];
    const float* w_hh0  = (const float*)d_in[6];
    const float* b_ih0  = (const float*)d_in[7];
    const float* b_hh0  = (const float*)d_in[8];
    const float* w_ih1  = (const float*)d_in[9];
    const float* w_hh1  = (const float*)d_in[10];
    const float* b_ih1  = (const float*)d_in[11];
    const float* b_hh1  = (const float*)d_in[12];
    const float* out_w  = (const float*)d_in[13];
    const float* out_b  = (const float*)d_in[14];
    float* out = (float*)d_out;

    // ---- workspace layout (256B aligned slabs) ----
    char* ws = (char*)d_ws;
    size_t off = 0;
    auto take = [&](size_t bytes) -> char* {
        char* p = ws + off;
        off = (off + bytes + 255) & ~(size_t)255;
        return p;
    };
    unsigned short* wb_inp = (unsigned short*)take((size_t)HD  * HD * 2);
    unsigned short* wb_ih0 = (unsigned short*)take((size_t)HD4 * HD * 2);
    unsigned short* wb_hh0 = (unsigned short*)take((size_t)HD4 * HD * 2);
    unsigned short* wb_ih1 = (unsigned short*)take((size_t)HD4 * HD * 2);
    unsigned short* wb_hh1 = (unsigned short*)take((size_t)HD4 * HD * 2);
    unsigned short* feat_b = (unsigned short*)take((size_t)TT * HD * 2);
    unsigned short* x_b    = (unsigned short*)take((size_t)TT * HD * 2);
    float*          gih0   = (float*)take((size_t)TT * HD4 * 4);
    float*          g0     = (float*)take((size_t)HD4 * 4);
    float*          g1     = (float*)take((size_t)HD4 * 4);
    float*          c1     = (float*)take((size_t)HD * 4);
    float*          c2     = (float*)take((size_t)HD * 4);
    float*          h1f    = (float*)take((size_t)HD * 4);
    float*          h2f    = (float*)take((size_t)HD * 4);
    unsigned short* h1b    = (unsigned short*)take((size_t)HD * 2);
    unsigned short* h2b    = (unsigned short*)take((size_t)HD * 2);
    if (off > ws_size) return;   // workspace too small: bail (deterministically)

    // ---- one-time-per-call weight conversion fp32 -> bf16 ----
    cvt_bf16_kernel<<<4096, 256, 0, stream>>>(inp_w, wb_inp, (size_t)HD  * HD);
    cvt_bf16_kernel<<<8192, 256, 0, stream>>>(w_ih0, wb_ih0, (size_t)HD4 * HD);
    cvt_bf16_kernel<<<8192, 256, 0, stream>>>(w_hh0, wb_hh0, (size_t)HD4 * HD);
    cvt_bf16_kernel<<<8192, 256, 0, stream>>>(w_ih1, wb_ih1, (size_t)HD4 * HD);
    cvt_bf16_kernel<<<8192, 256, 0, stream>>>(w_hh1, wb_hh1, (size_t)HD4 * HD);

    // ---- time-parallel frontend ----
    conv_pool_kernel<<<(TT * 3 * 32 * 32) / 256, 256, 0, stream>>>(inp, conv_w,
                                                                   conv_b, feat_b);
    // X(bf16) = feat @ inp_w^T + inp_b   : M=64, N=3072, K=3072
    gemm_bf16_kernel<<<dim3(HD / 64, TT / 16), 128, 0, stream>>>(
        feat_b, HD, wb_inp, HD, inp_b, nullptr, nullptr, x_b, HD, HD);
    // Gih0(f32) = X @ w_ih0^T + b_ih0 + b_hh0 : M=64, N=12288, K=3072
    gemm_bf16_kernel<<<dim3(HD4 / 64, TT / 16), 128, 0, stream>>>(
        x_b, HD, wb_ih0, HD, b_ih0, b_hh0, gih0, nullptr, HD4, HD);

    init_state_kernel<<<(HD + 255) / 256, 256, 0, stream>>>(c1, c2, h1f, h2f, h1b, h2b);

    // ---- sequential recurrence ----
    for (int t = 0; t < TT; ++t) {
        // g0 = Gih0[t] + h1 @ w_hh0^T   (biases already folded into Gih0)
        matvec_single_kernel<<<HD4 / 64, 128, 0, stream>>>(
            h1b, wb_hh0, gih0 + (size_t)t * HD4, g0, HD4, HD);
        lstm_cell_kernel<<<(HD + 255) / 256, 256, 0, stream>>>(g0, c1, h1f, h1b);
        // g1 = h1 @ w_ih1^T + h2 @ w_hh1^T + b_ih1 + b_hh1
        matvec_dual_kernel<<<HD4 / 64, 128, 0, stream>>>(
            h1b, wb_ih1, h2b, wb_hh1, b_ih1, b_hh1, g1, HD4, HD);
        lstm_cell_kernel<<<(HD + 255) / 256, 256, 0, stream>>>(g1, c2, h2f, h2b);
        outdot_kernel<<<1, 256, 0, stream>>>(h2f, out_w, out_b, out, t);
    }
    (void)in_sizes; (void)n_in; (void)out_size;
}
